// SVDQuantLinear_52261162058321
// MI455X (gfx1250) — 8-bit matrix operand order repaired and checked by reading, compiled, not run on hardware
//
#include <hip/hip_runtime.h>
#include <stdint.h>

typedef __attribute__((ext_vector_type(8))) int   v8i;
typedef __attribute__((ext_vector_type(8))) float v8f;
typedef __attribute__((ext_vector_type(2))) float v2f;

#define T_DIM 256
#define K_DIM 3072
#define N_DIM 3072
#define R_DIM 32
#define QGRP  64
#define NGRP  (K_DIM / QGRP)
#define TT    (T_DIM / 16)
#define NT    (N_DIM / 16)
#define BLKB  1024

#define OFF_A8  0
#define SZ_A8   (TT * NGRP * BLKB)
#define OFF_W8  (OFF_A8 + SZ_A8)
#define SZ_W8   (NT * NGRP * BLKB)
#define OFF_AS  (OFF_W8 + SZ_W8)
#define SZ_AS   (NGRP * T_DIM * 4)
#define OFF_XD  (OFF_AS + SZ_AS)
#define SZ_XD   (T_DIM * R_DIM * 4)
#define OFF_WST (OFF_XD + SZ_XD)
#define SZ_WST  (NGRP * N_DIM * 4)

__global__ void svdq_quant_pack_act(const float* __restrict__ x,
                                    int8_t* __restrict__ a8,
                                    float* __restrict__ ast) {
    int tid = blockIdx.x * blockDim.x + threadIdx.x;
    if (tid >= T_DIM * NGRP) return;
    int g = tid % NGRP, t = tid / NGRP;
    const float* xp = x + (size_t)t * K_DIM + g * QGRP;
    float amax = 0.f;
    for (int i = 0; i < QGRP; ++i) amax = fmaxf(amax, fabsf(xp[i]));
    float s = fmaxf(amax * (1.0f / 7.0f), 1e-8f);
    ast[g * T_DIM + t] = s;
    float inv = 1.0f / s;
    int8_t* blk = a8 + ((size_t)(t >> 4) * NGRP + g) * BLKB;
    int m = t & 15;
    for (int k = 0; k < QGRP; ++k) {
        float q = rintf(xp[k] * inv);
        q = fminf(fmaxf(q, -8.f), 7.f);
        int half = (k >> 4) & 1;
        int d = ((k & 15) >> 2) + ((k >= 32) ? 4 : 0);
        int j = k & 3;
        blk[(half * 16 + m) * 32 + d * 4 + j] = (int8_t)(int)q;
    }
}

__global__ void svdq_pack_weights(const int* __restrict__ qw,
                                  const float* __restrict__ wsc,
                                  int8_t* __restrict__ w8,
                                  float* __restrict__ wst) {
    int tid = blockIdx.x * blockDim.x + threadIdx.x;
    if (tid >= N_DIM * NGRP) return;
    int g = tid % NGRP, n = tid / NGRP;
    wst[g * N_DIM + n] = wsc[(size_t)n * NGRP + g];
    const int* wp = qw + (size_t)n * K_DIM + g * QGRP;
    int8_t* blk = w8 + ((size_t)(n >> 4) * NGRP + g) * BLKB;
    int c = n & 15;
    for (int k = 0; k < QGRP; ++k) {
        int half = (k >> 4) & 1;
        int d = ((k & 15) >> 2) + ((k >= 32) ? 4 : 0);
        int j = k & 3;
        blk[(half * 16 + c) * 32 + d * 4 + j] = (int8_t)(wp[k] - 8);
    }
}

__global__ void svdq_lora_down(const float* __restrict__ x,
                               const float* __restrict__ ld,
                               float* __restrict__ xd) {
    int tid = blockIdx.x * blockDim.x + threadIdx.x;
    if (tid >= T_DIM * R_DIM) return;
    int r = tid % R_DIM, t = tid / R_DIM;
    const float4* xp = (const float4*)(x + (size_t)t * K_DIM);
    const float4* lp = (const float4*)(ld + (size_t)r * K_DIM);
    float acc = 0.f;
    for (int i = 0; i < K_DIM / 4; ++i) {
        float4 a = xp[i], b = lp[i];
        acc += a.x * b.x + a.y * b.y + a.z * b.z + a.w * b.w;
    }
    xd[t * R_DIM + r] = acc;
}

__global__ void __launch_bounds__(128)
svdq_w4a4_wmma_gemm(const int8_t* __restrict__ a8,
                    const int8_t* __restrict__ w8,
                    const float* __restrict__ ast,
                    const float* __restrict__ wst,
                    const float* __restrict__ xd,
                    const float* __restrict__ lora_up,
                    const float* __restrict__ bias,
                    float* __restrict__ out) {
    int wave = threadIdx.x >> 5;
    int lane = threadIdx.x & 31;
    int tile = blockIdx.x * 4 + wave;
    if (tile >= (TT / 2) * (NT / 2)) return;
    int nn2 = tile % (NT / 2);
    int tt2 = tile / (NT / 2);
    int half = lane >> 4;
    int nc = lane & 15;

    int tA0 = tt2 * 2, tA1 = tA0 + 1;
    int nB0 = nn2 * 2, nB1 = nB0 + 1;
    int n0 = nB0 * 16 + nc, n1 = nB1 * 16 + nc;
    int t0 = tA0 * 16 + half * 8;
    int t1 = tA1 * 16 + half * 8;

    const int8_t* ap0 = a8 + (size_t)(tA0 * NGRP) * BLKB + lane * 32;
    const int8_t* ap1 = a8 + (size_t)(tA1 * NGRP) * BLKB + lane * 32;
    const int8_t* bp0 = w8 + (size_t)(nB0 * NGRP) * BLKB + lane * 32;
    const int8_t* bp1 = w8 + (size_t)(nB1 * NGRP) * BLKB + lane * 32;

    v8f acc00 = {}, acc01 = {}, acc10 = {}, acc11 = {};
    const v8i Z = {};

    v8i A0 = *(const v8i*)(ap0);
    v8i A1 = *(const v8i*)(ap1);
    v8i B0 = *(const v8i*)(bp0);
    v8i B1 = *(const v8i*)(bp1);
    v8i cP00 = __builtin_amdgcn_wmma_i32_16x16x64_iu8(true, A0, true, B0, Z, false, false);
    v8i cP01 = __builtin_amdgcn_wmma_i32_16x16x64_iu8(true, A0, true, B1, Z, false, false);
    v8i cP10 = __builtin_amdgcn_wmma_i32_16x16x64_iu8(true, A1, true, B0, Z, false, false);
    v8i cP11 = __builtin_amdgcn_wmma_i32_16x16x64_iu8(true, A1, true, B1, Z, false, false);
    v8f cs0 = *(const v8f*)(ast + t0);
    v8f cs1 = *(const v8f*)(ast + t1);
    float cw0 = wst[n0];
    float cw1 = wst[n1];

    for (int g = 1; g < NGRP; ++g) {
        v8i nA0 = *(const v8i*)(ap0 + (size_t)g * BLKB);
        v8i nA1 = *(const v8i*)(ap1 + (size_t)g * BLKB);
        v8i nB0 = *(const v8i*)(bp0 + (size_t)g * BLKB);
        v8i nB1 = *(const v8i*)(bp1 + (size_t)g * BLKB);
        if (g + 1 < NGRP) {
            __builtin_prefetch((const void*)(ap0 + (size_t)(g + 1) * BLKB), 0, 1);
            __builtin_prefetch((const void*)(ap1 + (size_t)(g + 1) * BLKB), 0, 1);
            __builtin_prefetch((const void*)(bp0 + (size_t)(g + 1) * BLKB), 0, 1);
            __builtin_prefetch((const void*)(bp1 + (size_t)(g + 1) * BLKB), 0, 1);
        }
        v8i P00 = __builtin_amdgcn_wmma_i32_16x16x64_iu8(true, nA0, true, nB0, Z, false, false);
        v8i P01 = __builtin_amdgcn_wmma_i32_16x16x64_iu8(true, nA0, true, nB1, Z, false, false);
        v8i P10 = __builtin_amdgcn_wmma_i32_16x16x64_iu8(true, nA1, true, nB0, Z, false, false);
        v8i P11 = __builtin_amdgcn_wmma_i32_16x16x64_iu8(true, nA1, true, nB1, Z, false, false);

#pragma unroll
        for (int i = 0; i < 8; ++i) {
            acc00[i] += (float)cP00[i] * (cs0[i] * cw0);
            acc01[i] += (float)cP01[i] * (cs0[i] * cw1);
            acc10[i] += (float)cP10[i] * (cs1[i] * cw0);
            acc11[i] += (float)cP11[i] * (cs1[i] * cw1);
        }

        cP00 = P00; cP01 = P01; cP10 = P10; cP11 = P11;
        cs0 = *(const v8f*)(ast + g * T_DIM + t0);
        cs1 = *(const v8f*)(ast + g * T_DIM + t1);
        cw0 = wst[g * N_DIM + n0];
        cw1 = wst[g * N_DIM + n1];
    }

#pragma unroll
    for (int i = 0; i < 8; ++i) {
        acc00[i] += (float)cP00[i] * (cs0[i] * cw0);
        acc01[i] += (float)cP01[i] * (cs0[i] * cw1);
        acc10[i] += (float)cP10[i] * (cs1[i] * cw0);
        acc11[i] += (float)cP11[i] * (cs1[i] * cw1);
    }

#pragma unroll
    for (int kc = 0; kc < R_DIM / 4; ++kc) {
        int ka = kc * 4 + 2 * half;
        int kb = kc * 4 + half;
        v2f a0, a1, b0, b1;
        a0.x = xd[(tA0 * 16 + nc) * R_DIM + ka];
        a0.y = xd[(tA0 * 16 + nc) * R_DIM + ka + 1];
        a1.x = xd[(tA1 * 16 + nc) * R_DIM + ka];
        a1.y = xd[(tA1 * 16 + nc) * R_DIM + ka + 1];
        b0.x = lora_up[(size_t)n0 * R_DIM + kb];
        b0.y = lora_up[(size_t)n0 * R_DIM + kb + 2];
        b1.x = lora_up[(size_t)n1 * R_DIM + kb];
        b1.y = lora_up[(size_t)n1 * R_DIM + kb + 2];
        acc00 = __builtin_amdgcn_wmma_f32_16x16x4_f32(false, a0, false, b0,
                                                      (short)0, acc00, false, false);
        acc01 = __builtin_amdgcn_wmma_f32_16x16x4_f32(false, a0, false, b1,
                                                      (short)0, acc01, false, false);
        acc10 = __builtin_amdgcn_wmma_f32_16x16x4_f32(false, a1, false, b0,
                                                      (short)0, acc10, false, false);
        acc11 = __builtin_amdgcn_wmma_f32_16x16x4_f32(false, a1, false, b1,
                                                      (short)0, acc11, false, false);
    }

    float bv0 = bias[n0];
    float bv1 = bias[n1];
#pragma unroll
    for (int i = 0; i < 8; ++i) {
        out[(size_t)(t0 + i) * N_DIM + n0] = acc00[i] + bv0;
        out[(size_t)(t0 + i) * N_DIM + n1] = acc01[i] + bv1;
        out[(size_t)(t1 + i) * N_DIM + n0] = acc10[i] + bv0;
        out[(size_t)(t1 + i) * N_DIM + n1] = acc11[i] + bv1;
    }
}

extern "C" void kernel_launch(void* const* d_in, const int* in_sizes, int n_in,
                              void* d_out, int out_size, void* d_ws, size_t ws_size,
                              hipStream_t stream) {
    const float* x    = (const float*)d_in[0];
    const int*   qw   = (const int*)d_in[1];
    const float* wsc  = (const float*)d_in[2];
    const float* ld   = (const float*)d_in[3];
    const float* lu   = (const float*)d_in[4];
    const float* bias = (const float*)d_in[5];
    float* out = (float*)d_out;

    char* ws = (char*)d_ws;
    int8_t* a8  = (int8_t*)(ws + OFF_A8);
    int8_t* w8  = (int8_t*)(ws + OFF_W8);
    float*  ast = (float*)(ws + OFF_AS);
    float*  xd  = (float*)(ws + OFF_XD);
    float*  wst = (float*)(ws + OFF_WST);

    svdq_quant_pack_act<<<(T_DIM * NGRP + 255) / 256, 256, 0, stream>>>(x, a8, ast);
    svdq_pack_weights<<<(N_DIM * NGRP + 255) / 256, 256, 0, stream>>>(qw, wsc, w8, wst);
    svdq_lora_down<<<(T_DIM * R_DIM + 255) / 256, 256, 0, stream>>>(x, ld, xd);
    int tiles = (TT / 2) * (NT / 2);
    svdq_w4a4_wmma_gemm<<<tiles / 4, 128, 0, stream>>>(a8, w8, ast, wst,
                                                       xd, lu, bias, out);
}
